// MeshVAE_41815801594349
// MI455X (gfx1250) — matrix operand order repaired and checked by reading, not compiled, not run on hardware
//
#include <hip/hip_runtime.h>

typedef __attribute__((ext_vector_type(16))) _Float16 v16h;
typedef __attribute__((ext_vector_type(8)))  float    v8f;

// ---------------------------------------------------------------------------
// Weight conversion f32 -> f16 with K padding (pad with zeros)
// dst is (dout x dinPad) row-major f16, src is (dout x din) row-major f32
// ---------------------------------------------------------------------------
__global__ void cvt_pad_kernel(const float* __restrict__ src, _Float16* __restrict__ dst,
                               int dout, int din, int dinPad) {
    int total = dout * dinPad;
    for (int idx = blockIdx.x * blockDim.x + threadIdx.x; idx < total;
         idx += gridDim.x * blockDim.x) {
        int j = idx % dinPad;
        int i = idx / dinPad;
        float v = (j < din) ? src[i * din + j] : 0.0f;
        dst[idx] = (_Float16)v;
    }
}

// ---------------------------------------------------------------------------
// CSR row pointers from sorted row[] array (every vertex has a self loop)
// ---------------------------------------------------------------------------
__global__ void rowptr_kernel(const int* __restrict__ row, int* __restrict__ row_ptr,
                              int nnz, int V) {
    int e = blockIdx.x * blockDim.x + threadIdx.x;
    if (e < nnz) {
        if (e == 0 || row[e] != row[e - 1]) row_ptr[row[e]] = e;
    }
    if (e == 0) row_ptr[V] = nnz;
}

// ---------------------------------------------------------------------------
// x (B,V,3) f32 -> padded f16 (N x 32)
// ---------------------------------------------------------------------------
__global__ void prep_x_kernel(const float* __restrict__ x, _Float16* __restrict__ H,
                              int total /* N*32 */) {
    for (int idx = blockIdx.x * blockDim.x + threadIdx.x; idx < total;
         idx += gridDim.x * blockDim.x) {
        int j = idx & 31;
        int n = idx >> 5;
        float v = (j < 3) ? x[n * 3 + j] : 0.0f;
        H[idx] = (_Float16)v;
    }
}

// ---------------------------------------------------------------------------
// WMMA fragment loaders per CDNA5 ISA 7.12.2 (wave32)
//   A: 16x32 f16. lane%16 = M row; VGPR v (halfs 2v,2v+1):
//      v<4 : K = 2v + (lane>=16 ? 8 : 0)
//      v>=4: K = 16 + 2(v-4) + (lane>=16 ? 8 : 0)
//   B: 32x16 f16. lane%16 = N col; VGPR v: same K mapping as A
// Each lane touches two contiguous 16-byte runs -> compiler emits b128 loads.
// ---------------------------------------------------------------------------
__device__ __forceinline__ void load_a_frag(const _Float16* __restrict__ A, int K,
                                            int row0, int kbase, int lane, v16h& a) {
    int m  = lane & 15;
    int hi = lane >> 4;
    const _Float16* ap = A + (size_t)(row0 + m) * K + kbase;
#pragma unroll
    for (int v = 0; v < 8; ++v) {
        int k = (v < 4) ? (2 * v + (hi ? 8 : 0)) : (16 + 2 * (v - 4) + (hi ? 8 : 0));
        a[2 * v]     = ap[k];
        a[2 * v + 1] = ap[k + 1];
    }
}

__device__ __forceinline__ void load_b_frag(const _Float16* __restrict__ W, int K,
                                            int col0, int kbase, int lane, v16h& b) {
    int n  = lane & 15;
    int hi = lane >> 4;
    const _Float16* wp = W + (size_t)(col0 + n) * K + kbase;
#pragma unroll
    for (int v = 0; v < 8; ++v) {
        int k = (v < 4) ? (2 * v + (hi ? 8 : 0)) : (16 + 2 * (v - 4) + (hi ? 8 : 0));
        b[2 * v]     = wp[k];
        b[2 * v + 1] = wp[k + 1];
    }
}

// ---------------------------------------------------------------------------
// Strip GEMM: C(N x dout) f16 = act( A(N x K) f16 @ W(dout x K)^T + bias )
//   - one wave owns a full 16-row strip (all TN = dout/16 column tiles),
//     so each A fragment is loaded exactly once per K step (8x/4x less
//     A traffic than tile-per-wave on this bandwidth-bound chip)
//   - weights staged in LDS once per block; B fragments come from ds_load
//   - accumulation in f32 via v_wmma_f32_16x16x32_f16
// ---------------------------------------------------------------------------
template <int TN, int RELU>
__global__ __launch_bounds__(256) void gemm_wmma_strip(const _Float16* __restrict__ A,
                                                       const _Float16* __restrict__ W,
                                                       const float* __restrict__ bias,
                                                       _Float16* __restrict__ C,
                                                       int Nrows, int K) {
    constexpr int DOUT = TN * 16;
    __shared__ _Float16 wlds[128 * 128];   // 32 KB max (dout<=128, K<=128)

    // cooperative 16B-wide stage of W into LDS
    {
        const uint4* src = (const uint4*)W;
        uint4*       dst = (uint4*)wlds;
        int n16 = (DOUT * K) >> 3;         // halves / 8 per 16B
        for (int i = threadIdx.x; i < n16; i += blockDim.x) dst[i] = src[i];
    }
    __syncthreads();

    int lane   = threadIdx.x & 31;
    int wave   = (blockIdx.x * blockDim.x + threadIdx.x) >> 5;
    int nWaves = (gridDim.x * blockDim.x) >> 5;
    int tilesM = Nrows >> 4;

    for (int tm = wave; tm < tilesM; tm += nWaves) {
        v8f acc[TN];
#pragma unroll
        for (int tn = 0; tn < TN; ++tn) acc[tn] = (v8f){};

        for (int kb = 0; kb < K; kb += 32) {
            v16h a;
            load_a_frag(A, K, tm * 16, kb, lane, a);
#pragma unroll
            for (int tn = 0; tn < TN; ++tn) {
                v16h b;
                load_b_frag(wlds, K, tn * 16, kb, lane, b);
                // (neg_a, A, neg_b, B, c_mod, C, reuse_a, reuse_b)
                acc[tn] = __builtin_amdgcn_wmma_f32_16x16x32_f16(
                    false, a, false, b, (short)0, acc[tn], false, false);
            }
        }

        int n  = lane & 15;
        int hi = lane >> 4;
#pragma unroll
        for (int tn = 0; tn < TN; ++tn) {
            float bv = bias[tn * 16 + n];
#pragma unroll
            for (int r = 0; r < 8; ++r) {
                int rowi = tm * 16 + r + (hi ? 8 : 0);
                float vv = acc[tn][r] + bv;
                if (RELU) vv = fmaxf(vv, 0.0f);
                C[(size_t)rowi * DOUT + tn * 16 + n] = (_Float16)vv;
            }
        }
    }
}

// ---------------------------------------------------------------------------
// SpMM + ReLU: out[b,i,c] = relu( sum_e vals[e] * in[b,col[e],c] )  (f16 io)
// ---------------------------------------------------------------------------
__global__ void spmm_relu_kernel(const _Float16* __restrict__ Yin,
                                 const int* __restrict__ row_ptr,
                                 const int* __restrict__ col,
                                 const float* __restrict__ vals,
                                 _Float16* __restrict__ Hout,
                                 int V, int C, int total /* B*V*C */) {
    for (int idx = blockIdx.x * blockDim.x + threadIdx.x; idx < total;
         idx += gridDim.x * blockDim.x) {
        int c   = idx % C;
        int rem = idx / C;
        int i   = rem % V;
        int b   = rem / V;
        const _Float16* base = Yin + (size_t)b * V * C + c;
        float s = 0.0f;
        int e1 = row_ptr[i + 1];
        for (int e = row_ptr[i]; e < e1; ++e)
            s += vals[e] * (float)base[(size_t)col[e] * C];
        Hout[idx] = (_Float16)fmaxf(s, 0.0f);
    }
}

// ---------------------------------------------------------------------------
// Fused SpMM + ReLU + mean over vertices: g[b,c] = mean_i relu(spmm(...)[b,i,c])
// One block per (b,c) pair.
// ---------------------------------------------------------------------------
__global__ void spmm_mean_kernel(const _Float16* __restrict__ Yin,
                                 const int* __restrict__ row_ptr,
                                 const int* __restrict__ col,
                                 const float* __restrict__ vals,
                                 float* __restrict__ g, int V, int C) {
    int b = blockIdx.x / C;
    int c = blockIdx.x % C;
    const _Float16* base = Yin + (size_t)b * V * C + c;
    float s = 0.0f;
    for (int i = threadIdx.x; i < V; i += blockDim.x) {
        float acc = 0.0f;
        int e1 = row_ptr[i + 1];
        for (int e = row_ptr[i]; e < e1; ++e)
            acc += vals[e] * (float)base[(size_t)col[e] * C];
        s += fmaxf(acc, 0.0f);
    }
    __shared__ float red[256];
    red[threadIdx.x] = s;
    __syncthreads();
    for (int off = 128; off > 0; off >>= 1) {
        if ((int)threadIdx.x < off) red[threadIdx.x] += red[threadIdx.x + off];
        __syncthreads();
    }
    if (threadIdx.x == 0) g[b * C + c] = red[0] / (float)V;
}

// ---------------------------------------------------------------------------
// VAE head: feat = relu(g@Wfc^T+bfc); mu, log_var (clipped); z = mu+eps*exp(lv/2)
// Single block. g:(16,128), feat:(16,256) in LDS, mu/lv:(16,32)
// ---------------------------------------------------------------------------
__global__ void head_kernel(const float* __restrict__ g,
                            const float* __restrict__ Wfc, const float* __restrict__ bfc,
                            const float* __restrict__ Wmu, const float* __restrict__ bmu,
                            const float* __restrict__ Wlv, const float* __restrict__ blv,
                            const float* __restrict__ eps,
                            float* __restrict__ z,
                            float* __restrict__ out_mu, float* __restrict__ out_lv,
                            int Bn) {
    __shared__ float feat[16 * 256];
    int nf = Bn * 256;
    for (int idx = threadIdx.x; idx < nf; idx += blockDim.x) {
        int b = idx / 256, j = idx % 256;
        const float* gb = g + b * 128;
        const float* wj = Wfc + j * 128;
        float s = bfc[j];
        for (int k = 0; k < 128; ++k) s += gb[k] * wj[k];
        feat[idx] = fmaxf(s, 0.0f);
    }
    __syncthreads();
    int nz = Bn * 32;
    for (int idx = threadIdx.x; idx < nz; idx += blockDim.x) {
        int b = idx / 32, j = idx % 32;
        const float* fb = feat + b * 256;
        const float* wm = Wmu + j * 256;
        const float* wl = Wlv + j * 256;
        float mu = bmu[j], lv = blv[j];
        for (int k = 0; k < 256; ++k) { mu += fb[k] * wm[k]; lv += fb[k] * wl[k]; }
        lv = fminf(fmaxf(lv, -20.0f), 20.0f);
        out_mu[idx] = mu;
        out_lv[idx] = lv;
        z[idx] = mu + eps[idx] * __expf(0.5f * lv);
    }
}

// ---------------------------------------------------------------------------
// Decoder input: hd0 (N x 64) f16 = [ z(32) | template(3) | zeros(29) ]
// ---------------------------------------------------------------------------
__global__ void prep_hd_kernel(const float* __restrict__ z, const float* __restrict__ tmpl,
                               _Float16* __restrict__ H, int V, int total /* N*64 */) {
    for (int idx = blockIdx.x * blockDim.x + threadIdx.x; idx < total;
         idx += gridDim.x * blockDim.x) {
        int j = idx & 63;
        int n = idx >> 6;
        int v = n % V;
        int b = n / V;
        float val = 0.0f;
        if (j < 32)      val = z[b * 32 + j];
        else if (j < 35) val = tmpl[v * 3 + (j - 32)];
        H[idx] = (_Float16)val;
    }
}

// ---------------------------------------------------------------------------
// Final projection 64->3 + template add: recon = template + hd3 @ Wd4^T + bd4
// ---------------------------------------------------------------------------
__global__ void d4_recon_kernel(const _Float16* __restrict__ H,
                                const float* __restrict__ Wd4, const float* __restrict__ bd4,
                                const float* __restrict__ tmpl, float* __restrict__ recon,
                                int V, int total /* N*3 */) {
    for (int idx = blockIdx.x * blockDim.x + threadIdx.x; idx < total;
         idx += gridDim.x * blockDim.x) {
        int o = idx % 3;
        int n = idx / 3;
        int v = n % V;
        const _Float16* h = H + (size_t)n * 64;
        const float* w = Wd4 + o * 64;
        float s = bd4[o];
#pragma unroll 4
        for (int k = 0; k < 64; ++k) s += (float)h[k] * w[k];
        recon[idx] = tmpl[v * 3 + o] + s;
    }
}

// ---------------------------------------------------------------------------
// Host side
// ---------------------------------------------------------------------------
extern "C" void kernel_launch(void* const* d_in, const int* in_sizes, int n_in,
                              void* d_out, int out_size, void* d_ws, size_t ws_size,
                              hipStream_t stream) {
    const float* x    = (const float*)d_in[0];
    const float* eps  = (const float*)d_in[1];
    const int*   row  = (const int*)d_in[2];
    const int*   col  = (const int*)d_in[3];
    const float* vals = (const float*)d_in[4];
    const float* tmpl = (const float*)d_in[5];
    const float* W1  = (const float*)d_in[6];   const float* b1  = (const float*)d_in[7];
    const float* W2  = (const float*)d_in[8];   const float* b2  = (const float*)d_in[9];
    const float* W3  = (const float*)d_in[10];  const float* b3  = (const float*)d_in[11];
    const float* Wfc = (const float*)d_in[12];  const float* bfc = (const float*)d_in[13];
    const float* Wmu = (const float*)d_in[14];  const float* bmu = (const float*)d_in[15];
    const float* Wlv = (const float*)d_in[16];  const float* blv = (const float*)d_in[17];
    const float* Wd1 = (const float*)d_in[18];  const float* bd1 = (const float*)d_in[19];
    const float* Wd2 = (const float*)d_in[20];  const float* bd2 = (const float*)d_in[21];
    const float* Wd3 = (const float*)d_in[22];  const float* bd3 = (const float*)d_in[23];
    const float* Wd4 = (const float*)d_in[24];  const float* bd4 = (const float*)d_in[25];

    const int Bn  = in_sizes[1] / 32;     // 16
    const int nnz = in_sizes[2];
    const int V   = in_sizes[5] / 3;      // 10242
    const size_t N = (size_t)Bn * V;      // 163872, multiple of 16

    // workspace carve-up (256B aligned)
    char* ws = (char*)d_ws;
    size_t off = 0;
    auto carve = [&](size_t bytes) -> void* {
        void* p = ws + off;
        off = (off + bytes + 255) & ~(size_t)255;
        return p;
    };
    _Float16* Ha   = (_Float16*)carve(N * 128 * sizeof(_Float16));
    _Float16* Hb   = (_Float16*)carve(N * 128 * sizeof(_Float16));
    _Float16* W1p  = (_Float16*)carve(64 * 32 * sizeof(_Float16));
    _Float16* W2h  = (_Float16*)carve(64 * 64 * sizeof(_Float16));
    _Float16* W3h  = (_Float16*)carve(128 * 64 * sizeof(_Float16));
    _Float16* Wd1p = (_Float16*)carve(128 * 64 * sizeof(_Float16));
    _Float16* Wd2h = (_Float16*)carve(128 * 128 * sizeof(_Float16));
    _Float16* Wd3h = (_Float16*)carve(64 * 128 * sizeof(_Float16));
    float*    gbuf = (float*)carve((size_t)Bn * 128 * sizeof(float));
    float*    zbuf = (float*)carve((size_t)Bn * 32 * sizeof(float));
    int*      rp   = (int*)carve((size_t)(V + 1) * sizeof(int));

    float* recon  = (float*)d_out;                       // (B,V,3)
    float* out_mu = recon + N * 3;                       // (B,32)
    float* out_lv = out_mu + (size_t)Bn * 32;            // (B,32)

    const int T = 256;
    auto nb = [](long total, int t) { return (int)((total + t - 1) / t); };

    // --- weight prep (f32 -> f16, K padded to multiple of 32) ---
    cvt_pad_kernel<<<nb(64 * 32, T), T, 0, stream>>>(W1, W1p, 64, 3, 32);
    cvt_pad_kernel<<<nb(64 * 64, T), T, 0, stream>>>(W2, W2h, 64, 64, 64);
    cvt_pad_kernel<<<nb(128 * 64, T), T, 0, stream>>>(W3, W3h, 128, 64, 64);
    cvt_pad_kernel<<<nb(128 * 64, T), T, 0, stream>>>(Wd1, Wd1p, 128, 35, 64);
    cvt_pad_kernel<<<nb(128 * 128, T), T, 0, stream>>>(Wd2, Wd2h, 128, 128, 128);
    cvt_pad_kernel<<<nb(64 * 128, T), T, 0, stream>>>(Wd3, Wd3h, 64, 128, 128);
    rowptr_kernel<<<nb(nnz, T), T, 0, stream>>>(row, rp, nnz, V);

    // --- encoder ---
    prep_x_kernel<<<nb((long)N * 32, T), T, 0, stream>>>(x, Ha, (int)(N * 32));

    auto gemm = [&](const _Float16* A, const _Float16* Wp, const float* bias,
                    _Float16* Cp, int K, int dout, int relu) {
        int tilesM = (int)(N / 16);
        int blocks = (tilesM + 7) / 8;   // 8 waves (256 threads) per block
        if (dout == 64) {
            if (relu) gemm_wmma_strip<4, 1><<<blocks, 256, 0, stream>>>(A, Wp, bias, Cp, (int)N, K);
            else      gemm_wmma_strip<4, 0><<<blocks, 256, 0, stream>>>(A, Wp, bias, Cp, (int)N, K);
        } else {
            if (relu) gemm_wmma_strip<8, 1><<<blocks, 256, 0, stream>>>(A, Wp, bias, Cp, (int)N, K);
            else      gemm_wmma_strip<8, 0><<<blocks, 256, 0, stream>>>(A, Wp, bias, Cp, (int)N, K);
        }
    };

    gemm(Ha, W1p, b1, Hb, 32, 64, 0);
    spmm_relu_kernel<<<nb((long)N * 64, T), T, 0, stream>>>(Hb, rp, col, vals, Ha, V, 64,
                                                            (int)(N * 64));
    gemm(Ha, W2h, b2, Hb, 64, 64, 0);
    spmm_relu_kernel<<<nb((long)N * 64, T), T, 0, stream>>>(Hb, rp, col, vals, Ha, V, 64,
                                                            (int)(N * 64));
    gemm(Ha, W3h, b3, Hb, 64, 128, 0);
    // fused SpMM + ReLU + vertex-mean (h3 never materialized)
    spmm_mean_kernel<<<Bn * 128, 256, 0, stream>>>(Hb, rp, col, vals, gbuf, V, 128);

    // --- VAE head ---
    head_kernel<<<1, 256, 0, stream>>>(gbuf, Wfc, bfc, Wmu, bmu, Wlv, blv, eps,
                                       zbuf, out_mu, out_lv, Bn);

    // --- decoder ---
    prep_hd_kernel<<<nb((long)N * 64, T), T, 0, stream>>>(zbuf, tmpl, Ha, V, (int)(N * 64));
    gemm(Ha, Wd1p, bd1, Hb, 64, 128, 1);
    gemm(Hb, Wd2h, bd2, Ha, 128, 128, 1);
    gemm(Ha, Wd3h, bd3, Hb, 128, 64, 1);
    d4_recon_kernel<<<nb((long)N * 3, T), T, 0, stream>>>(Hb, Wd4, bd4, tmpl, recon, V,
                                                          (int)(N * 3));
    (void)n_in; (void)out_size; (void)ws_size;
}
